// MMoE_670014899152
// MI455X (gfx1250) — matrix operand order repaired and checked by reading, not compiled, not run on hardware
//
#include <hip/hip_runtime.h>

// ---------------------------------------------------------------------------
// MMoE fused kernel for MI455X (gfx1250), wave32, WMMA bf16 path.
// B=16384, D=512, U=128, E=16, T=4.  N = U*E + T*E = 2048 + 64 = 2112 cols.
// ---------------------------------------------------------------------------

typedef __attribute__((ext_vector_type(16))) __bf16 v16bf;
typedef __attribute__((ext_vector_type(8)))  float  v8f;

typedef unsigned short u16;
typedef unsigned int   u32;

#define BDIM   16384
#define DDIM   512
#define UDIM   128
#define EDIM   16
#define TDIM   4
#define NCOLS  2112           // 2048 expert + 64 gate
#define NTILES 132            // NCOLS / 16 (even)
#define KBLKS  16             // DDIM / 32
#define ROWT   1024           // BDIM / 16 row tiles
#define CPITCH 2116           // LDS row pitch for the 16 x 2112 f32 stage
#define FRAGSZ (KBLKS * 32 * 16)   // u16 elems per 16-wide column tile (16 KB)

// fragment blob sizes (in u16 elements)
#define AFRAG_PER_RB (KBLKS * 32 * 16)            // 8192 u16 = 16 KB per 16-row tile
#define AFRAG_TOTAL  (ROWT * AFRAG_PER_RB)        // 8,388,608 u16 = 16 MB
#define WFRAG_TOTAL  (NTILES * FRAGSZ)            // 1,081,344 u16 ≈ 2.2 MB

#define SMEM_BYTES (AFRAG_PER_RB * 2 + 16 * CPITCH * 4)   // 16384 + 135424 = 151808

__device__ __forceinline__ u16 f2bf(float f) {
  union { float f; u32 u; } v; v.f = f;
  u32 u = v.u;
  u += 0x7FFFu + ((u >> 16) & 1u);   // round-to-nearest-even
  return (u16)(u >> 16);
}

struct Frag32B { uint4 lo, hi; };

__device__ __forceinline__ v16bf load_frag(const u16* p) {
  Frag32B t;
  t.lo = *(const uint4*)(p);
  t.hi = *(const uint4*)(p + 8);
  return __builtin_bit_cast(v16bf, t);
}

// ---------------------------------------------------------------------------
// pack_a: x[B,D] f32 -> bf16 pre-swizzled into 16x32 A-fragment layout.
// Afrag[rb][kb][lane][j]:
//   row   = rb*16 + (lane & 15)
//   khalf = lane >> 4
//   k     = kb*32 + khalf*8 + j          (j in 0..7,  VGPR0..3)
//   k     = kb*32 + 16 + khalf*8 + (j-8) (j in 8..15, VGPR4..7)
// ---------------------------------------------------------------------------
__global__ void pack_a(const float* __restrict__ x, u16* __restrict__ af) {
  u32 idx = blockIdx.x * blockDim.x + threadIdx.x;   // one per output u16
  u32 j    = idx & 15u;
  u32 lane = (idx >> 4) & 31u;
  u32 kb   = (idx >> 9) & 15u;
  u32 rb   = idx >> 13;
  u32 row   = rb * 16 + (lane & 15u);
  u32 khalf = lane >> 4;
  u32 k = kb * 32 + khalf * 8 + ((j < 8) ? j : (16 + (j - 8)));
  af[idx] = f2bf(x[row * DDIM + k]);
}

// ---------------------------------------------------------------------------
// pack_w: combined weight [K=512, N=2112] -> bf16 32x16 B-fragment layout.
//   n < 2048 : expert_kernels[d,u,e] flat = k*2048 + n
//   n >= 2048: gate_kernels[t,d,e]   flat = (t*512 + k)*16 + e
// Wfrag[ct][kb][lane][j]:
//   n     = ct*16 + (lane & 15)
//   khalf = lane >> 4
//   k     = kb*32 + khalf*8 + j           (j in 0..7),  kb*32 + 16 + khalf*8 + (j-8) (j in 8..15)
// ---------------------------------------------------------------------------
__global__ void pack_w(const float* __restrict__ we,
                       const float* __restrict__ wg,
                       u16* __restrict__ wf) {
  u32 idx = blockIdx.x * blockDim.x + threadIdx.x;
  u32 j    = idx & 15u;
  u32 lane = (idx >> 4) & 31u;
  u32 kb   = (idx >> 9) & 15u;
  u32 ct   = idx >> 13;
  u32 n     = ct * 16 + (lane & 15u);
  u32 khalf = lane >> 4;
  u32 k = kb * 32 + khalf * 8 + ((j < 8) ? j : (16 + (j - 8)));
  float src;
  if (n < (u32)(UDIM * EDIM)) {
    src = we[k * (UDIM * EDIM) + n];
  } else {
    u32 nn = n - UDIM * EDIM;
    u32 t = nn >> 4, e = nn & 15u;
    src = wg[(t * DDIM + k) * EDIM + e];
  }
  wf[idx] = f2bf(src);
}

// ---------------------------------------------------------------------------
// Fused main kernel: 1 WG (8 waves, 256 thr) per 16 rows of x.
//   stage 1: copy pre-swizzled A tile (16 KB) to LDS
//   stage 2: each wave: two column tiles per iteration (2 independent WMMA
//            accumulation chains sharing one A fragment), +bias, relu for
//            expert cols, spill to LDS f32 stage [16][2116]
//   stage 3: softmax over E per (t,row) in LDS (64 threads)
//   stage 4: gated combine -> 4 task outputs
// ---------------------------------------------------------------------------
__device__ __forceinline__ void store_tile(float* sC, const v8f& acc, u32 ct,
                                           u32 lane,
                                           const float* __restrict__ ebias,
                                           const float* __restrict__ gbias) {
  const u32 col   = ct * 16 + (lane & 15u);
  const u32 rbase = (lane >> 4) * 8;
  const bool isExpert = (col < (u32)(UDIM * EDIM));
  const float bias = isExpert ? ebias[col] : gbias[col - UDIM * EDIM];
  #pragma unroll
  for (u32 r = 0; r < 8; ++r) {
    float v = acc[r] + bias;
    if (isExpert && v < 0.0f) v = 0.0f;
    sC[(rbase + r) * CPITCH + col] = v;
  }
}

__global__ void mmoe_fused(const u16* __restrict__ af,
                           const u16* __restrict__ wf,
                           const float* __restrict__ ebias,
                           const float* __restrict__ gbias,
                           float* __restrict__ out) {
  extern __shared__ char smem[];
  u16*   sA = (u16*)smem;                              // [KBLKS][32][16] u16
  float* sC = (float*)(smem + AFRAG_PER_RB * 2);       // [16][CPITCH] f32

  const u32 tid  = threadIdx.x;
  const u32 wave = tid >> 5;
  const u32 lane = tid & 31u;
  const u32 rb   = blockIdx.x;

  // ---- stage 1: A tile -> LDS (16 KB, coalesced b128 copies) ----
  {
    const uint4* src = (const uint4*)(af + (size_t)rb * AFRAG_PER_RB);
    uint4* dst = (uint4*)sA;
    #pragma unroll
    for (u32 i = tid; i < AFRAG_PER_RB / 8; i += 256) {  // 1024 uint4
      dst[i] = src[i];
    }
  }
  __syncthreads();

  // ---- stage 2: WMMA, two column tiles per wave iteration ----
  // wave w owns tile pairs {2w+16i, 2w+1+16i}; NTILES is even so ct1 < NTILES.
  for (u32 base = wave * 2; base < NTILES; base += 16) {
    const u32 ct0 = base, ct1 = base + 1;
    const u16* wp0 = wf + (size_t)ct0 * FRAGSZ + lane * 16;
    const u16* wp1 = wf + (size_t)ct1 * FRAGSZ + lane * 16;
    if (base + 16 < NTILES) {
      __builtin_prefetch(wf + (size_t)(base + 16) * FRAGSZ + lane * 16, 0, 1);
      __builtin_prefetch(wf + (size_t)(base + 17) * FRAGSZ + lane * 16, 0, 1);
    }
    v8f acc0 = {}, acc1 = {};
    #pragma unroll
    for (u32 kb = 0; kb < KBLKS; ++kb) {
      v16bf a  = load_frag(sA + kb * 512 + lane * 16);
      v16bf b0 = load_frag(wp0 + kb * 512);
      v16bf b1 = load_frag(wp1 + kb * 512);
      acc0 = __builtin_amdgcn_wmma_f32_16x16x32_bf16(
          false, a, false, b0, (short)0, acc0, false, false);
      acc1 = __builtin_amdgcn_wmma_f32_16x16x32_bf16(
          false, a, false, b1, (short)0, acc1, false, false);
    }
    store_tile(sC, acc0, ct0, lane, ebias, gbias);
    store_tile(sC, acc1, ct1, lane, ebias, gbias);
  }
  __syncthreads();

  // ---- stage 3: per-(t,row) softmax over E=16 gate logits ----
  if (tid < 64) {
    const u32 t = tid >> 4, row = tid & 15u;
    float* g = &sC[row * CPITCH + UDIM * EDIM + t * EDIM];
    float m = g[0];
    #pragma unroll
    for (int e = 1; e < EDIM; ++e) m = fmaxf(m, g[e]);
    float s = 0.0f;
    #pragma unroll
    for (int e = 0; e < EDIM; ++e) { float v = __expf(g[e] - m); g[e] = v; s += v; }
    const float inv = __frcp_rn(s);
    #pragma unroll
    for (int e = 0; e < EDIM; ++e) g[e] *= inv;
  }
  __syncthreads();

  // ---- stage 4: outs[t,b,u] = sum_e expert[b,u,e] * gate[t,b,e] ----
  {
    const u32 row = tid >> 4;          // 0..15
    const u32 ub  = tid & 15u;         // 0..15
    const float* gz = &sC[row * CPITCH + UDIM * EDIM];
    const u32 gr = rb * 16 + row;
    #pragma unroll
    for (u32 i = 0; i < 8; ++i) {
      const u32 u = ub + 16 * i;
      const float* ev = &sC[row * CPITCH + u * EDIM];
      float a0 = 0.f, a1 = 0.f, a2 = 0.f, a3 = 0.f;
      #pragma unroll
      for (int e = 0; e < EDIM; ++e) {
        const float v = ev[e];
        a0 = fmaf(v, gz[0 * EDIM + e], a0);
        a1 = fmaf(v, gz[1 * EDIM + e], a1);
        a2 = fmaf(v, gz[2 * EDIM + e], a2);
        a3 = fmaf(v, gz[3 * EDIM + e], a3);
      }
      const size_t base = (size_t)gr * UDIM + u;
      out[0 * (size_t)BDIM * UDIM + base] = a0;
      out[1 * (size_t)BDIM * UDIM + base] = a1;
      out[2 * (size_t)BDIM * UDIM + base] = a2;
      out[3 * (size_t)BDIM * UDIM + base] = a3;
    }
  }
}

// ---------------------------------------------------------------------------
extern "C" void kernel_launch(void* const* d_in, const int* in_sizes, int n_in,
                              void* d_out, int out_size, void* d_ws, size_t ws_size,
                              hipStream_t stream) {
  const float* x  = (const float*)d_in[0];   // [B, D]
  const float* we = (const float*)d_in[1];   // [D, U, E]
  const float* eb = (const float*)d_in[2];   // [U, E]
  const float* wg = (const float*)d_in[3];   // [T, D, E]
  const float* gb = (const float*)d_in[4];   // [T, E]
  float* out = (float*)d_out;                // T * [B, U], concatenated

  u16* af  = (u16*)d_ws;                     // 16 MB
  u16* wfr = (u16*)((char*)d_ws + (size_t)AFRAG_TOTAL * 2);  // 2.2 MB

  // pack x into A-fragment order (bf16)
  pack_a<<<AFRAG_TOTAL / 256, 256, 0, stream>>>(x, af);
  // pack expert+gate weights into B-fragment order (bf16)
  pack_w<<<WFRAG_TOTAL / 256, 256, 0, stream>>>(we, wg, wfr);

  // fused GEMM + relu + softmax + gated combine
  hipFuncSetAttribute((const void*)mmoe_fused,
                      hipFuncAttributeMaxDynamicSharedMemorySize, SMEM_BYTES);
  mmoe_fused<<<ROWT, 256, SMEM_BYTES, stream>>>(af, wfr, eb, gb, out);
}
